// MultiQueryAttentionLayerV2_28527172780135
// MI455X (gfx1250) — matrix operand order repaired and checked by reading, not compiled, not run on hardware
//
#include <hip/hip_runtime.h>
#include <hip/hip_bf16.h>

typedef __bf16 v16bf __attribute__((ext_vector_type(16)));
typedef float  v8f   __attribute__((ext_vector_type(8)));

#define BB 4
#define NN 2048
#define MM 2048
#define DD 512
#define HH 8
#define KK 64
#define VV 64
#define HKK (HH*KK)   // 512
#define HVV (HH*VV)   // 512
#define L2E 1.44269504088896340736f

// ---------- helpers ----------

__device__ __forceinline__ unsigned short f2bf(float f) {
    unsigned int u = __float_as_uint(f);
    u += 0x7FFFu + ((u >> 16) & 1u);          // round-to-nearest-even
    return (unsigned short)(u >> 16);
}

union FragU { uint4 u[2]; v16bf v; };

// B fragment (32x16, K x N): lane = output column (lc), 16 contiguous K values
// starting at (lane<16 ? 0 : 16). p points at that run. (same k order as load_a_frag)
__device__ __forceinline__ v16bf load_b_frag(const __bf16* p) {
    FragU f;
    f.u[0] = *(const uint4*)(p);
    f.u[1] = *(const uint4*)(p + 8);
    return f.v;
}

// A fragment (16x32, M x K): lane holds row lane%16; 16 contiguous K values
// K in {0..15} (lanes 0-15) / {16..31} (lanes 16-31).
// p points at that run.
__device__ __forceinline__ v16bf load_a_frag(const __bf16* p) {
    FragU f;
    f.u[0] = *(const uint4*)(p);
    f.u[1] = *(const uint4*)(p + 8);
    return f.v;
}

__device__ __forceinline__ v8f zero8() {
    v8f c;
#pragma unroll
    for (int i = 0; i < 8; ++i) c[i] = 0.0f;
    return c;
}

__device__ __forceinline__ v8f wmma_bf16(v16bf a, v16bf b, v8f c) {
    return __builtin_amdgcn_wmma_f32_16x16x32_bf16(false, a, false, b,
                                                   (short)0, c, false, false);
}

// 16x64 output stripe = A[16 x kdim] (row-major, stride lda) times four
// 16-column tiles of B^T[64 x kdim] (row j = output column j, contiguous k).
// A fragment is loaded once per k-chunk and reused across the 4 WMMAs.
__device__ __forceinline__ void gemm16x64(const __bf16* A, int lda,
                                          const __bf16* Bt, int ldb,
                                          int kdim, int lane, v8f acc[4]) {
    const int half = lane >> 4;
    const int lc   = lane & 15;
    const __bf16* arow = A  + (size_t)lc * lda + (half ? 16 : 0);
    const __bf16* brow = Bt + (size_t)lc * ldb + (half ? 16 : 0);
#pragma unroll
    for (int t = 0; t < 4; ++t) acc[t] = zero8();
    for (int k0 = 0; k0 < kdim; k0 += 32) {
        v16bf a = load_a_frag(arow + k0);
#pragma unroll
        for (int t = 0; t < 4; ++t) {
            v16bf b = load_b_frag(brow + (size_t)t * 16 * ldb + k0);
            acc[t] = wmma_bf16(a, b, acc[t]);
        }
    }
}

// C tile (8 VGPR f32): element r -> (row = r + 8*(lane>=16), col = lane&15)
__device__ __forceinline__ void store_tile_bf16(unsigned short* dst, int ldd,
                                                v8f c, int lane) {
    const int half = lane >> 4;
    const int lc   = lane & 15;
#pragma unroll
    for (int r = 0; r < 8; ++r)
        dst[(size_t)(r + 8 * half) * ldd + lc] = f2bf(c[r]);
}

// ---------- conversion kernels ----------

// 8 elements per thread: two b128 loads, one packed b128 store.
__global__ void __launch_bounds__(256) k_cvt8(const float* __restrict__ src,
                                              unsigned short* __restrict__ dst,
                                              int n8) {
    int i = blockIdx.x * 256 + threadIdx.x;
    if (i < n8) {
        const float4* s = (const float4*)src + (size_t)i * 2;
        float4 a = s[0], b = s[1];
        union { unsigned short h[8]; uint4 u; } o;
        o.h[0] = f2bf(a.x); o.h[1] = f2bf(a.y);
        o.h[2] = f2bf(a.z); o.h[3] = f2bf(a.w);
        o.h[4] = f2bf(b.x); o.h[5] = f2bf(b.y);
        o.h[6] = f2bf(b.z); o.h[7] = f2bf(b.w);
        *((uint4*)dst + i) = o.u;
    }
}

// src [rows, cols] fp32 -> dst [cols, rows] bf16  (small weight matrices)
__global__ void __launch_bounds__(256) k_cvt_t(const float* __restrict__ src,
                                               unsigned short* __restrict__ dst,
                                               int rows, int cols) {
    int i = blockIdx.x * 256 + threadIdx.x;
    if (i < rows * cols) {
        int r = i / cols, c = i % cols;
        dst[(size_t)c * rows + r] = f2bf(src[i]);
    }
}

// ---------- q projection: q[B*N, H*K] = x[B*N, D] @ Wq^T ----------
// Wq given as [H*K, D] row-major == B^T row-major. 512 row x 8 col-64 tiles.

__global__ void __launch_bounds__(256) k_qproj(const __bf16* __restrict__ xb,
                                               const __bf16* __restrict__ wq,
                                               unsigned short* __restrict__ qb) {
    const int lane = threadIdx.x & 31;
    const int gw   = (blockIdx.x << 3) + (threadIdx.x >> 5);
    const int ct   = gw & 7;    // HKK/64 col-super tiles
    const int rt   = gw >> 3;   // (B*N)/16 row tiles
    v8f acc[4];
    gemm16x64(xb + (size_t)rt * 16 * DD, DD,
              wq + (size_t)ct * 64 * DD, DD, DD, lane, acc);
#pragma unroll
    for (int t = 0; t < 4; ++t)
        store_tile_bf16(qb + (size_t)rt * 16 * HKK + ct * 64 + t * 16,
                        HKK, acc[t], lane);
}

// ---------- k/v projection ----------
// k[B*M, K] = m @ Wk   (WkT passed as [K, D])
// vT[B, V, M] = (m @ Wv)^T  (WvT passed as [V, D]); transposed store so the
// flash kernel's B-fragments for P@V are contiguous per lane.

__global__ void __launch_bounds__(256) k_kvproj(const __bf16* __restrict__ mb,
                                                const __bf16* __restrict__ kpT,
                                                const __bf16* __restrict__ vpT,
                                                unsigned short* __restrict__ kb,
                                                unsigned short* __restrict__ vtb) {
    const int lane = threadIdx.x & 31;
    const int gw   = (blockIdx.x << 3) + (threadIdx.x >> 5);
    const int ct   = gw & 1;    // 0 -> k (64 cols), 1 -> v (64 cols)
    const int rt   = gw >> 1;   // (B*M)/16 row tiles
    const int half = lane >> 4;
    const int lc   = lane & 15;
    const __bf16* A = mb + (size_t)rt * 16 * DD;
    v8f acc[4];
    if (ct == 0) {
        gemm16x64(A, DD, kpT, DD, DD, lane, acc);
#pragma unroll
        for (int t = 0; t < 4; ++t)
            store_tile_bf16(kb + (size_t)rt * 16 * KK + t * 16, KK, acc[t], lane);
    } else {
        gemm16x64(A, DD, vpT, DD, DD, lane, acc);
#pragma unroll
        for (int t = 0; t < 4; ++t) {
#pragma unroll
            for (int r = 0; r < 8; ++r) {
                int grow = rt * 16 + r + 8 * half;   // global row in [0, B*M)
                int b    = grow >> 11;               // / MM
                int mm   = grow & (MM - 1);
                vtb[((size_t)b * VV + t * 16 + lc) * MM + mm] = f2bf(acc[t][r]);
            }
        }
    }
}

// ---------- flash attention ----------
// One wave owns 16 query rows of one (b, h); streams M in chunks of 32.

__global__ void __launch_bounds__(256) k_flash(const __bf16* __restrict__ qb,
                                               const __bf16* __restrict__ kb,
                                               const __bf16* __restrict__ vtb,
                                               unsigned short* __restrict__ ob) {
    __shared__ __bf16 lds_p[8][16 * 32];   // per-wave P re-layout buffer
    const int wave = threadIdx.x >> 5;
    const int lane = threadIdx.x & 31;
    const int half = lane >> 4;
    const int lc   = lane & 15;
    const int gw   = (blockIdx.x << 3) + wave;   // 0 .. B*H*(N/16)-1
    const int ntile = gw & (NN / 16 - 1);
    const int bh    = gw >> 7;
    const int h     = bh & (HH - 1);
    const int b     = bh >> 3;
    const int n0    = ntile * 16;

    // Q fragments: rows n0..n0+15, head-dim K=64 -> two k=32 chunks
    const __bf16* qrow = qb + (size_t)(b * NN + n0 + lc) * HKK + h * KK + (half ? 16 : 0);
    v16bf qf0 = load_a_frag(qrow);
    v16bf qf1 = load_a_frag(qrow + 32);

    float rm[8], rl[8];
    v8f oacc[4];
#pragma unroll
    for (int r = 0; r < 8; ++r) { rm[r] = -1e30f; rl[r] = 0.0f; }
#pragma unroll
    for (int t = 0; t < 4; ++t) oacc[t] = zero8();

    const __bf16* kbase = kb  + (size_t)b * MM * KK;
    const __bf16* vbase = vtb + (size_t)b * VV * MM + (half ? 16 : 0);
    unsigned short* lp  = (unsigned short*)&lds_p[wave][0];
    const __bf16*   lpr = &lds_p[wave][0] + lc * 32 + (half ? 16 : 0);

    for (int m0 = 0; m0 < MM; m0 += 32) {
        // S = Q @ K^T : two 16x16 C tiles (cols m0..+15, m0+16..+31)
        const __bf16* k0r = kbase + (size_t)(m0 + lc) * KK + (half ? 16 : 0);
        const __bf16* k1r = kbase + (size_t)(m0 + 16 + lc) * KK + (half ? 16 : 0);
        v8f s0 = zero8(), s1 = zero8();
        s0 = wmma_bf16(qf0, load_b_frag(k0r), s0);
        s0 = wmma_bf16(qf1, load_b_frag(k0r + 32), s0);
        s1 = wmma_bf16(qf0, load_b_frag(k1r), s1);
        s1 = wmma_bf16(qf1, load_b_frag(k1r + 32), s1);

        // online softmax (per-row stats live per VGPR-index r, per lane-half)
#pragma unroll
        for (int r = 0; r < 8; ++r) {
            float v0 = s0[r] * 0.125f;            // 1/sqrt(K)
            float v1 = s1[r] * 0.125f;
            float cm = fmaxf(v0, v1);
            cm = fmaxf(cm, __shfl_xor(cm, 1));
            cm = fmaxf(cm, __shfl_xor(cm, 2));
            cm = fmaxf(cm, __shfl_xor(cm, 4));
            cm = fmaxf(cm, __shfl_xor(cm, 8));    // row max over 16 cols, within half
            float mn = fmaxf(rm[r], cm);
            float co = exp2f((rm[r] - mn) * L2E);
            rm[r] = mn;
            float p0 = exp2f((v0 - mn) * L2E);
            float p1 = exp2f((v1 - mn) * L2E);
            float cs = p0 + p1;
            cs += __shfl_xor(cs, 1);
            cs += __shfl_xor(cs, 2);
            cs += __shfl_xor(cs, 4);
            cs += __shfl_xor(cs, 8);
            rl[r] = rl[r] * co + cs;
            oacc[0][r] *= co; oacc[1][r] *= co;
            oacc[2][r] *= co; oacc[3][r] *= co;
            // C-layout -> row-major P tile in LDS (bf16)
            int row = r + 8 * half;
            lp[row * 32 + lc]      = f2bf(p0);
            lp[row * 32 + 16 + lc] = f2bf(p1);
        }
        asm volatile("s_wait_dscnt 0" ::: "memory");
        v16bf pa = load_a_frag(lpr);              // P as 16x32 A fragment

        // O += P @ V(32 x 64): four 16-col tiles, vT rows are contiguous in m
#pragma unroll
        for (int t = 0; t < 4; ++t) {
            const __bf16* vrow = vbase + (size_t)(t * 16 + lc) * MM + m0;
            oacc[t] = wmma_bf16(pa, load_b_frag(vrow), oacc[t]);
        }
    }

    // normalize and store o[B*N, H*V] (bf16) for the output projection
#pragma unroll
    for (int r = 0; r < 8; ++r) {
        float inv = 1.0f / rl[r];
        size_t base = (size_t)(b * NN + n0 + r + 8 * half) * HVV + h * VV + lc;
#pragma unroll
        for (int t = 0; t < 4; ++t)
            ob[base + t * 16] = f2bf(oacc[t][r] * inv);
    }
}

// ---------- output projection: out[B*N, D] = o[B*N, H*V] @ Wo' ----------
// Wo given as [D, H*V] row-major: row d = contiguous hv == B^T row-major.

__global__ void __launch_bounds__(256) k_outproj(const __bf16* __restrict__ ob,
                                                 const __bf16* __restrict__ wo,
                                                 float* __restrict__ out) {
    const int lane = threadIdx.x & 31;
    const int gw   = (blockIdx.x << 3) + (threadIdx.x >> 5);
    const int ct   = gw & 7;    // D/64 col-super tiles
    const int rt   = gw >> 3;   // (B*N)/16 row tiles
    v8f acc[4];
    gemm16x64(ob + (size_t)rt * 16 * HVV, HVV,
              wo + (size_t)ct * 64 * HVV, HVV, HVV, lane, acc);
    const int half = lane >> 4;
    const int lc   = lane & 15;
#pragma unroll
    for (int t = 0; t < 4; ++t)
#pragma unroll
        for (int r = 0; r < 8; ++r)
            out[(size_t)(rt * 16 + r + 8 * half) * DD + ct * 64 + t * 16 + lc] =
                acc[t][r];
}

// ---------- launcher ----------

extern "C" void kernel_launch(void* const* d_in, const int* in_sizes, int n_in,
                              void* d_out, int out_size, void* d_ws, size_t ws_size,
                              hipStream_t stream) {
    (void)in_sizes; (void)n_in; (void)out_size; (void)ws_size;
    const float* x  = (const float*)d_in[0];   // [B,N,D]
    const float* m  = (const float*)d_in[1];   // [B,M,D]
    const float* wq = (const float*)d_in[2];   // [H,K,D]
    const float* kp = (const float*)d_in[3];   // [D,K]
    const float* vp = (const float*)d_in[4];   // [D,V]
    const float* wo = (const float*)d_in[5];   // [D,H,V]

    char* ws = (char*)d_ws;
    size_t off = 0;
    auto alloc = [&](size_t bytes) { char* p = ws + off; off += bytes; return p; };
    unsigned short* xb   = (unsigned short*)alloc((size_t)BB * NN * DD * 2);
    unsigned short* mb   = (unsigned short*)alloc((size_t)BB * MM * DD * 2);
    unsigned short* wqb  = (unsigned short*)alloc((size_t)HKK * DD * 2);
    unsigned short* kptb = (unsigned short*)alloc((size_t)KK * DD * 2);
    unsigned short* vptb = (unsigned short*)alloc((size_t)VV * DD * 2);
    unsigned short* wob  = (unsigned short*)alloc((size_t)DD * HVV * 2);
    unsigned short* q_b  = (unsigned short*)alloc((size_t)BB * NN * HKK * 2);
    unsigned short* k_b  = (unsigned short*)alloc((size_t)BB * MM * KK * 2);
    unsigned short* vt_b = (unsigned short*)alloc((size_t)BB * VV * MM * 2);
    unsigned short* o_b  = (unsigned short*)alloc((size_t)BB * NN * HVV * 2);

    const int nXM8 = BB * NN * DD / 8;                  // 524,288 octets
    k_cvt8<<<(nXM8 + 255) / 256, 256, 0, stream>>>(x, xb, nXM8);
    k_cvt8<<<(nXM8 + 255) / 256, 256, 0, stream>>>(m, mb, nXM8);
    const int nW8 = HKK * DD / 8;                       // 32,768 octets
    k_cvt8<<<(nW8 + 255) / 256, 256, 0, stream>>>(wq, wqb, nW8);
    k_cvt8<<<(nW8 + 255) / 256, 256, 0, stream>>>(wo, wob, nW8);
    k_cvt_t<<<(DD * KK + 255) / 256, 256, 0, stream>>>(kp, kptb, DD, KK);
    k_cvt_t<<<(DD * VV + 255) / 256, 256, 0, stream>>>(vp, vptb, DD, VV);

    // q-proj: (B*N/16) * (HKK/64) = 512*8 stripes, 8 waves/block
    k_qproj<<<512, 256, 0, stream>>>((const __bf16*)xb, (const __bf16*)wqb, q_b);
    // k/v-proj: 512 row tiles * 2 (k|v) stripes
    k_kvproj<<<128, 256, 0, stream>>>((const __bf16*)mb, (const __bf16*)kptb,
                                      (const __bf16*)vptb, k_b, vt_b);
    // flash: B*H*(N/16) = 4096 waves
    k_flash<<<512, 256, 0, stream>>>((const __bf16*)q_b, (const __bf16*)k_b,
                                     (const __bf16*)vt_b, o_b);
    // out-proj: 512*8 stripes
    k_outproj<<<512, 256, 0, stream>>>((const __bf16*)o_b, (const __bf16*)wob,
                                       (float*)d_out);
}
